// SyntaxBasedDecoderLSTM_44040594653389
// MI455X (gfx1250) — matrix operand order repaired and checked by reading, not compiled, not run on hardware
//
#include <hip/hip_runtime.h>
#include <hip/hip_bf16.h>
#include <math.h>

// Problem dims
#define T_STEPS 128
#define S_LEN   1024
#define HDIM    2048
#define ATTD    512
#define AEMB    128
#define NEMB    32
#define RVD     256
#define TVD     32000
#define GH      8192      // 4*H (gate vector length)
#define NSPLIT  8         // split-K factor for recurrent mat-vecs
#define QSPLIT  8         // split-K for h1@Wq
#define CSPLIT  4         // split-K for sc@enc

typedef __attribute__((ext_vector_type(16))) __bf16 v16bf;
typedef __attribute__((ext_vector_type(8)))  __bf16 v8bf;
typedef __attribute__((ext_vector_type(4)))  __bf16 v4bf;
typedef __attribute__((ext_vector_type(8)))  float  v8f;

__device__ __forceinline__ float sigf(float x) { return 1.0f / (1.0f + __expf(-x)); }

// ---------------------------------------------------------------------------
// Zero the 4 contiguous state vectors (c0,h0,c1,h1)
__global__ void k_init_state(float* st) {
  int i = blockIdx.x * blockDim.x + threadIdx.x;
  if (i < 4 * HDIM) st[i] = 0.0f;
}

// ---------------------------------------------------------------------------
// xpre[t][j] = b0[j] + prev_emb_t . Wx0[0:128] + node_emb_t . Wx0[128:160]
// (teacher forcing: prev action is known ahead of time). Also emits action_type.
__global__ void k_precompute_xpre(const int* __restrict__ av, const int* __restrict__ ntp,
                                  const int* __restrict__ n2a,
                                  const float* __restrict__ rule_embed,
                                  const float* __restrict__ token_embed,
                                  const float* __restrict__ node_embed,
                                  const float* __restrict__ Wx0, const float* __restrict__ b0,
                                  float* __restrict__ xpre, float* __restrict__ out_at) {
  int j = blockIdx.x * blockDim.x + threadIdx.x;   // 0..8191
  int t = blockIdx.y;
  int nt = ntp[t];
  int at = n2a[nt];
  int pt = (t == 0) ? 0 : n2a[ntp[t - 1]];
  int pv = (t == 0) ? 0 : av[t - 1];
  const float* pe = (pt == 0 ? rule_embed : token_embed) + (size_t)pv * AEMB;
  const float* ne = node_embed + (size_t)nt * NEMB;
  float acc = b0[j];
#pragma unroll 8
  for (int i = 0; i < AEMB; ++i) acc += pe[i] * Wx0[(size_t)i * GH + j];
#pragma unroll 8
  for (int i = 0; i < NEMB; ++i) acc += ne[i] * Wx0[(size_t)(AEMB + i) * GH + j];
  xpre[(size_t)t * GH + j] = acc;
  if (j == 0) out_at[t] = (float)at;
}

// ---------------------------------------------------------------------------
// Generic deterministic partial reduce: out[i] = sum_p part[p][i]
__global__ void k_reduce_partials(const float* __restrict__ part, int nparts, int len,
                                  float* __restrict__ out) {
  int i = blockIdx.x * blockDim.x + threadIdx.x;
  if (i >= len) return;
  float s = 0.0f;
  for (int p = 0; p < nparts; ++p) s += part[(size_t)p * len + i];
  out[i] = s;
}

// h1q partial = h1[chunk] @ Wq[chunk]   [2048 x 512], split-K x8
__global__ void k_attn_query_part(const float* __restrict__ h1, const float* __restrict__ Wq,
                                  float* __restrict__ qpart) {
  int a = blockIdx.x * blockDim.x + threadIdx.x;   // 0..511
  int p = blockIdx.y;                              // 0..QSPLIT-1
  int base = p * (HDIM / QSPLIT);
  float acc = 0.0f;
#pragma unroll 8
  for (int h = base; h < base + HDIM / QSPLIT; ++h) acc += h1[h] * Wq[(size_t)h * ATTD + a];
  qpart[(size_t)p * ATTD + a] = acc;
}

// sc_raw[s] = mask[s] ? sum_a tanh(pk[s][a] + h1q[a]) * We[a] : -inf
__global__ void k_attn_energy(const float* __restrict__ pk, const float* __restrict__ h1q,
                              const float* __restrict__ We, const unsigned char* __restrict__ mask,
                              float* __restrict__ sc_raw) {
  __shared__ float red[256];
  int s = blockIdx.x, tid = threadIdx.x;
  float acc = 0.0f;
  for (int a = tid; a < ATTD; a += 256)
    acc += tanhf(pk[(size_t)s * ATTD + a] + h1q[a]) * We[a];
  red[tid] = acc; __syncthreads();
  for (int off = 128; off > 0; off >>= 1) {
    if (tid < off) red[tid] += red[tid + off];
    __syncthreads();
  }
  if (tid == 0) sc_raw[s] = mask[s] ? red[0] : -INFINITY;
}

// softmax over S=1024 (single block of 1024 threads); also writes scores output
__global__ void k_attn_softmax(const float* __restrict__ sc_raw, float* __restrict__ sc,
                               float* __restrict__ out_scores) {
  __shared__ float buf[S_LEN];
  int tid = threadIdx.x;
  float v = sc_raw[tid];
  buf[tid] = v; __syncthreads();
  for (int off = 512; off > 0; off >>= 1) {
    if (tid < off) buf[tid] = fmaxf(buf[tid], buf[tid + off]);
    __syncthreads();
  }
  float m = buf[0]; __syncthreads();
  float e = __expf(v - m);
  buf[tid] = e; __syncthreads();
  for (int off = 512; off > 0; off >>= 1) {
    if (tid < off) buf[tid] += buf[tid + off];
    __syncthreads();
  }
  float p = e / buf[0];
  sc[tid] = p;
  out_scores[tid] = p;
}

// ctx partial = sc[chunk] @ enc[chunk]   [1024 x 2048], split-K x4
__global__ void k_attn_context_part(const float* __restrict__ sc, const float* __restrict__ enc,
                                    float* __restrict__ cpart) {
  int h = blockIdx.x * blockDim.x + threadIdx.x;   // 0..2047
  int p = blockIdx.y;                              // 0..CSPLIT-1
  int base = p * (S_LEN / CSPLIT);
  float acc = 0.0f;
#pragma unroll 4
  for (int s = base; s < base + S_LEN / CSPLIT; ++s) acc += sc[s] * enc[(size_t)s * HDIM + h];
  cpart[(size_t)p * HDIM + h] = acc;
}

// ---------------------------------------------------------------------------
// Split-K partial of gates = addv + va@Wa + vb@Wb  (virtual K = 4096, 8 chunks of 512)
__global__ void k_dual_matvec_partial(const float* __restrict__ va, const float* __restrict__ Wa,
                                      const float* __restrict__ vb, const float* __restrict__ Wb,
                                      const float* __restrict__ addv, float* __restrict__ gpart) {
  int j = blockIdx.x * blockDim.x + threadIdx.x;  // 0..8191
  int p = blockIdx.y;                             // 0..NSPLIT-1
  const float* v = (p < NSPLIT / 2) ? va : vb;
  const float* W = (p < NSPLIT / 2) ? Wa : Wb;
  int base = (p & (NSPLIT / 2 - 1)) * (HDIM / (NSPLIT / 2));  // 0,512,1024,1536
  float acc = (p == 0) ? addv[j] : 0.0f;
#pragma unroll 8
  for (int i = base; i < base + HDIM / (NSPLIT / 2); ++i)
    acc += v[i] * W[(size_t)i * GH + j];
  gpart[(size_t)p * GH + j] = acc;
}

// Reduce split-K partials, apply flax LSTMCell pointwise math (gate order i,f,g,o)
__global__ void k_lstm_update(const float* __restrict__ gpart, float* __restrict__ c,
                              float* __restrict__ h, float* __restrict__ h_save) {
  int j = blockIdx.x * blockDim.x + threadIdx.x;
  if (j >= HDIM) return;
  float g4[4];
#pragma unroll
  for (int g = 0; g < 4; ++g) {
    float s = 0.0f;
#pragma unroll
    for (int p = 0; p < NSPLIT; ++p) s += gpart[(size_t)p * GH + g * HDIM + j];
    g4[g] = s;
  }
  float cn = sigf(g4[1]) * c[j] + sigf(g4[0]) * tanhf(g4[2]);
  float hn = sigf(g4[3]) * tanhf(cn);
  c[j] = cn;
  h[j] = hn;
  if (h_save) h_save[j] = hn;
}

// ---------------------------------------------------------------------------
// rule_logits = H1 @ Wr + br   [128 x 2048] @ [2048 x 256]
__global__ void k_rule_logits(const float* __restrict__ h1_all, const float* __restrict__ Wr,
                              const float* __restrict__ br, float* __restrict__ out_rule) {
  int r = threadIdx.x;   // 256
  int t = blockIdx.x;    // 128
  const float* h = h1_all + (size_t)t * HDIM;
  float acc = br[r];
#pragma unroll 8
  for (int k = 0; k < HDIM; ++k) acc += h[k] * Wr[(size_t)k * RVD + r];
  out_rule[(size_t)t * RVD + r] = acc;
}

// ---------------------------------------------------------------------------
// token_logits = H1 @ Wt + bt via bf16 WMMA, fp32 accumulate.
// Block = 256 threads = 8 waves; wave w owns M-tile w (rows 16w..16w+15);
// block owns a 64-wide N stripe (4 N-tiles per wave, A fragment reused 4x).
// A staged [row][k] (per-lane fragment = two contiguous 16B runs -> ds_load_b128);
// B staged n-major [n][k] (per-lane fragment = two contiguous 16B runs).
__global__ void k_token_wmma(const float* __restrict__ h1_all, const float* __restrict__ Wt,
                             const float* __restrict__ bt, float* __restrict__ out_tok) {
  __shared__ __bf16 As[T_STEPS * 32];   // 128 rows x 32 K  (8 KB)
  __shared__ __bf16 Bs[64 * 32];        // 64 n  x 32 K     (4 KB)
  const int tid  = threadIdx.x;
  const int wave = tid >> 5;            // 0..7 -> M tile
  const int lane = tid & 31;
  const int half = lane >> 4;
  const int l16  = lane & 15;
  const int n0   = blockIdx.x * 64;
  v8f acc[4] = {};
  for (int k0 = 0; k0 < HDIM; k0 += 32) {
    // stage A: float4 global read -> packed 4x bf16 (b64 LDS store)
    for (int idx = tid * 4; idx < T_STEPS * 32; idx += 256 * 4) {
      int r = idx >> 5, cc = idx & 31;
      const float4 f = *(const float4*)(h1_all + (size_t)r * HDIM + k0 + cc);
      v4bf p4; p4[0] = (__bf16)f.x; p4[1] = (__bf16)f.y; p4[2] = (__bf16)f.z; p4[3] = (__bf16)f.w;
      *(v4bf*)&As[idx] = p4;
    }
    // stage B transposed: element (k, n) from coalesced float4 rows -> Bs[n*32 + k]
    for (int idx = tid * 4; idx < 32 * 64; idx += 256 * 4) {
      int k = idx >> 6, n = idx & 63;
      const float4 f = *(const float4*)(Wt + (size_t)(k0 + k) * TVD + n0 + n);
      Bs[(n + 0) * 32 + k] = (__bf16)f.x;
      Bs[(n + 1) * 32 + k] = (__bf16)f.y;
      Bs[(n + 2) * 32 + k] = (__bf16)f.z;
      Bs[(n + 3) * 32 + k] = (__bf16)f.w;
    }
    __syncthreads();
    // A fragment: lane holds row M, k in [half*8, half*8+8) then [16+half*8, ...)
    int row = (wave << 4) + l16;
    v8bf alo = *(const v8bf*)&As[row * 32 + (half << 3)];
    v8bf ahi = *(const v8bf*)&As[row * 32 + 16 + (half << 3)];
    v16bf a = __builtin_shufflevector(alo, ahi, 0, 1, 2, 3, 4, 5, 6, 7,
                                      8, 9, 10, 11, 12, 13, 14, 15);
#pragma unroll
    for (int nt = 0; nt < 4; ++nt) {
      // B fragment: lane holds col N, k in [half*8, half*8+8) then [16+half*8, ...)
      int cl = nt * 16 + l16;
      v8bf blo = *(const v8bf*)&Bs[cl * 32 + (half << 3)];
      v8bf bhi = *(const v8bf*)&Bs[cl * 32 + 16 + (half << 3)];
      v16bf b = __builtin_shufflevector(blo, bhi, 0, 1, 2, 3, 4, 5, 6, 7,
                                        8, 9, 10, 11, 12, 13, 14, 15);
      acc[nt] = __builtin_amdgcn_wmma_f32_16x16x32_bf16(false, a, false, b,
                                                        (short)0, acc[nt], false, false);
    }
    __syncthreads();
  }
  // C/D layout: VGPR r, lanes 0-15 -> M=r, lanes 16-31 -> M=r+8; N = l16
#pragma unroll
  for (int nt = 0; nt < 4; ++nt) {
    int col = n0 + nt * 16 + l16;
    float bias = bt[col];
#pragma unroll
    for (int r = 0; r < 8; ++r) {
      int m = (wave << 4) + r + (half << 3);
      out_tok[(size_t)m * TVD + col] = acc[nt][r] + bias;
    }
  }
}

// ---------------------------------------------------------------------------
// argmax(softmax(x)) == argmax(x); first-occurrence tie-break like jnp.argmax.
__global__ void k_predict(const float* __restrict__ out_rule, const float* __restrict__ out_tok,
                          const int* __restrict__ ntp, const int* __restrict__ n2a,
                          const unsigned char* __restrict__ valid, float* __restrict__ out_pred) {
  __shared__ float sval[256];
  __shared__ int   sidx[256];
  int t = blockIdx.x, tid = threadIdx.x;
  int nt = ntp[t];
  int at = n2a[nt];
  float rv = valid[(size_t)nt * RVD + tid] ? out_rule[(size_t)t * RVD + tid] : -INFINITY;
  sval[tid] = rv; sidx[tid] = tid; __syncthreads();
  for (int off = 128; off > 0; off >>= 1) {
    if (tid < off) {
      float v2 = sval[tid + off]; int i2 = sidx[tid + off];
      if (v2 > sval[tid] || (v2 == sval[tid] && i2 < sidx[tid])) { sval[tid] = v2; sidx[tid] = i2; }
    }
    __syncthreads();
  }
  int rule_idx = sidx[0];
  __syncthreads();
  float best = -INFINITY; int bidx = 0;
  for (int k = tid; k < TVD; k += 256) {
    float v = out_tok[(size_t)t * TVD + k];
    if (v > best) { best = v; bidx = k; }
  }
  sval[tid] = best; sidx[tid] = bidx; __syncthreads();
  for (int off = 128; off > 0; off >>= 1) {
    if (tid < off) {
      float v2 = sval[tid + off]; int i2 = sidx[tid + off];
      if (v2 > sval[tid] || (v2 == sval[tid] && i2 < sidx[tid])) { sval[tid] = v2; sidx[tid] = i2; }
    }
    __syncthreads();
  }
  if (tid == 0) out_pred[t] = (float)((at != 0) ? sidx[0] : rule_idx);
}

// ---------------------------------------------------------------------------
extern "C" void kernel_launch(void* const* d_in, const int* in_sizes, int n_in,
                              void* d_out, int out_size, void* d_ws, size_t ws_size,
                              hipStream_t stream) {
  (void)in_sizes; (void)n_in; (void)out_size; (void)ws_size;
  const int*   av    = (const int*)d_in[0];
  const int*   ntp   = (const int*)d_in[1];
  const float* enc   = (const float*)d_in[2];
  const float* pk    = (const float*)d_in[3];
  const unsigned char* mask  = (const unsigned char*)d_in[4];
  const int*   n2a   = (const int*)d_in[5];
  const unsigned char* valid = (const unsigned char*)d_in[6];
  const float* rule_embed  = (const float*)d_in[7];
  const float* token_embed = (const float*)d_in[8];
  const float* node_embed  = (const float*)d_in[9];
  const float* Wq  = (const float*)d_in[10];
  const float* We  = (const float*)d_in[11];
  const float* Wx0 = (const float*)d_in[12];
  const float* Wh0 = (const float*)d_in[13];
  const float* b0  = (const float*)d_in[14];
  const float* Wx1 = (const float*)d_in[15];
  const float* Wh1 = (const float*)d_in[16];
  const float* b1  = (const float*)d_in[17];
  const float* Wr  = (const float*)d_in[18];
  const float* br  = (const float*)d_in[19];
  const float* Wt  = (const float*)d_in[20];
  const float* bt  = (const float*)d_in[21];

  // Output layout (reference tuple order, flattened as float)
  float* out      = (float*)d_out;
  float* out_rule = out;                                   // T*RV
  float* out_tok  = out_rule + (size_t)T_STEPS * RVD;      // T*TV
  float* out_at   = out_tok  + (size_t)T_STEPS * TVD;      // T
  float* out_pred = out_at + T_STEPS;                      // T
  float* out_sc   = out_pred + T_STEPS;                    // T*S

  // Workspace (~5.7 MB of floats)
  float* ws = (float*)d_ws;
  float* h1_all = ws;  ws += (size_t)T_STEPS * HDIM;       // 1 MB
  float* xpre   = ws;  ws += (size_t)T_STEPS * GH;         // 4 MB
  float* c0 = ws; ws += HDIM;   // c0,h0,c1,h1 contiguous (zeroed together)
  float* h0 = ws; ws += HDIM;
  float* c1 = ws; ws += HDIM;
  float* h1 = ws; ws += HDIM;
  float* h1q    = ws; ws += ATTD;
  float* qpart  = ws; ws += (size_t)QSPLIT * ATTD;
  float* sc_raw = ws; ws += S_LEN;
  float* sc     = ws; ws += S_LEN;
  float* ctx    = ws; ws += HDIM;
  float* cpart  = ws; ws += (size_t)CSPLIT * HDIM;
  float* gpart  = ws; ws += (size_t)NSPLIT * GH;           // 256 KB

  k_init_state<<<(4 * HDIM + 255) / 256, 256, 0, stream>>>(c0);
  k_precompute_xpre<<<dim3(GH / 256, T_STEPS), 256, 0, stream>>>(
      av, ntp, n2a, rule_embed, token_embed, node_embed, Wx0, b0, xpre, out_at);

  for (int t = 0; t < T_STEPS; ++t) {
    k_attn_query_part<<<dim3(ATTD / 256, QSPLIT), 256, 0, stream>>>(h1, Wq, qpart);
    k_reduce_partials<<<ATTD / 256, 256, 0, stream>>>(qpart, QSPLIT, ATTD, h1q);
    k_attn_energy <<<S_LEN, 256, 0, stream>>>(pk, h1q, We, mask, sc_raw);
    k_attn_softmax<<<1, S_LEN, 0, stream>>>(sc_raw, sc, out_sc + (size_t)t * S_LEN);
    k_attn_context_part<<<dim3(HDIM / 256, CSPLIT), 256, 0, stream>>>(sc, enc, cpart);
    k_reduce_partials<<<HDIM / 256, 256, 0, stream>>>(cpart, CSPLIT, HDIM, ctx);
    // LSTM layer 0: gates = xpre[t] + ctx@Wx0[160:2208] + h0@Wh0
    k_dual_matvec_partial<<<dim3(GH / 256, NSPLIT), 256, 0, stream>>>(
        ctx, Wx0 + (size_t)(AEMB + NEMB) * GH, h0, Wh0, xpre + (size_t)t * GH, gpart);
    k_lstm_update<<<HDIM / 256, 256, 0, stream>>>(gpart, c0, h0, (float*)nullptr);
    // LSTM layer 1: gates = b1 + h0@Wx1 + h1@Wh1
    k_dual_matvec_partial<<<dim3(GH / 256, NSPLIT), 256, 0, stream>>>(
        h0, Wx1, h1, Wh1, b1, gpart);
    k_lstm_update<<<HDIM / 256, 256, 0, stream>>>(gpart, c1, h1, h1_all + (size_t)t * HDIM);
  }

  // Deferred batched projections (weight matrices read exactly once)
  k_rule_logits<<<T_STEPS, RVD, 0, stream>>>(h1_all, Wr, br, out_rule);
  k_token_wmma <<<TVD / 64, 256, 0, stream>>>(h1_all, Wt, bt, out_tok);
  k_predict    <<<T_STEPS, 256, 0, stream>>>(out_rule, out_tok, ntp, n2a, valid, out_pred);
}
